// Attention_69827578298511
// MI455X (gfx1250) — compile-verified, matrix operand order checked by reading, not run on hardware
//
#include <hip/hip_runtime.h>

// ---------------- problem constants (B=32, S=2048, H=512) ----------------
#define BD   32
#define SD   2048
#define HD   512
#define PS   520          // padded LDS row stride in ushorts (260 dwords -> bank spread)

typedef __attribute__((ext_vector_type(16))) __bf16 v16bf;
typedef __attribute__((ext_vector_type(8)))  float  v8f;

__device__ __forceinline__ unsigned pack_bf16(float a, float b) {
    unsigned ua = __builtin_bit_cast(unsigned, a);
    unsigned ub = __builtin_bit_cast(unsigned, b);
    ua = (ua + 0x7FFFu + ((ua >> 16) & 1u)) >> 16;   // round-to-nearest-even
    ub = (ub + 0x7FFFu + ((ub >> 16) & 1u)) >> 16;
    return ua | (ub << 16);
}

union FragU {
    struct { uint4 lo, hi; } u;   // 2 x 16B = 16 bf16
    v16bf v;
};

// ---------- kernel 1: W_e -> bf16, packed in WMMA B-fragment order ----------
// Block (kk, nt) = 16 N x 32 K tile stored as 1KB: lane l (0..31) owns 32B:
//   chunk0 = We[n = nt*16 + (l&15)][kk*32 + (l>>4)*8 + 0..7]
//   chunk1 = We[n                 ][kk*32 + 16 + (l>>4)*8 + 0..7]
// grid: 16 kk * 32 nt = 512 blocks; thread tid = l*8 + c2 converts one f32 pair.
__global__ __launch_bounds__(256) void we_convert(const float* __restrict__ W_att,
                                                  unsigned* __restrict__ We_p) {
    const int blk = blockIdx.x;           // kk*32 + nt
    const int nt  = blk & 31;
    const int kk  = blk >> 5;
    const int tid = threadIdx.x;
    const int l   = tid >> 3;
    const int c2  = tid & 7;
    const int n   = (nt << 4) + (l & 15);
    const int k   = (kk << 5) + ((c2 >> 2) << 4) + ((l >> 4) << 3) + ((c2 & 3) << 1);
    const float* src = W_att + (size_t)n * (2 * HD) + HD + k;
    We_p[(blk << 8) + (l << 3) + ((c2 >> 2) << 2) + (c2 & 3)] = pack_bf16(src[0], src[1]);
}

// ---------- kernel 2: q[b,n] = hidden[b,:]·W_att[n,0:H] + b_att[n] ----------
__global__ __launch_bounds__(256) void q_kernel(const float* __restrict__ hidden,
                                                const float* __restrict__ W_att,
                                                const float* __restrict__ b_att,
                                                float* __restrict__ q) {
    int j = blockIdx.x * 256 + threadIdx.x;    // 0 .. B*H-1
    int b = j >> 9;
    int n = j & (HD - 1);
    const float4* hr = (const float4*)(hidden + (size_t)b * HD);
    const float4* wr = (const float4*)(W_att + (size_t)n * (2 * HD));
    float acc = 0.f;
#pragma unroll 4
    for (int h4 = 0; h4 < HD / 4; ++h4) {
        float4 hv = hr[h4], wv = wr[h4];
        acc += hv.x * wv.x + hv.y * wv.y + hv.z * wv.z + hv.w * wv.w;
    }
    q[j] = acc + b_att[n];
}

// ---------- kernel 3: fused bf16-WMMA GEMM + tanh + dot(v_w) ----------
// grid = (B*S)/64 workgroups, 256 threads (8 wave32). Workgroup owns 64 rows
// of one batch. Wave w owns N columns [w*64, w*64+64) (4 n-tiles) x 4 m-tiles.
// Each B fragment is loaded once and reused by 4 WMMAs (4x less W_e traffic).
__global__ __launch_bounds__(256) void attn_gemm(const float* __restrict__ enc,
                                                 const unsigned short* __restrict__ We_b,
                                                 const float* __restrict__ q,
                                                 const float* __restrict__ v_w,
                                                 const float* __restrict__ v_b,
                                                 float* __restrict__ out_scores) {
    __shared__ unsigned short As[64 * PS];   // ~65 KB bf16 A tile (padded rows)
    __shared__ float sred[64];

    const int tile = blockIdx.x;
    const int b    = tile >> 5;              // 2048/64 = 32 tiles per batch
    const int s0   = (tile & 31) << 6;
    const int tid  = threadIdx.x;
    const float* encBase = enc + ((size_t)b * SD + s0) * HD;

    // phase 1: cooperative fp32 -> bf16 conversion into LDS (row r, pair tid)
    unsigned* AsU = (unsigned*)As;
#pragma unroll 8
    for (int row = 0; row < 64; ++row) {
        const float2 f = ((const float2*)(encBase + row * HD))[tid];
        AsU[row * (PS / 2) + tid] = pack_bf16(f.x, f.y);
    }
    if (tid < 64) sred[tid] = 0.f;
    __syncthreads();

    const int lane = tid & 31;
    const int wv   = tid >> 5;       // wave id 0..7
    const int hl   = lane >> 4;      // half-lane select (K halves)
    const int ll   = lane & 15;      // M (for A) / N (for B,D) index

    v8f acc[4][4] = {};              // [m-tile][n-tile]

    // packed B: byte offset = kk*32768 + wv*4096 + t*1024 + lane*32
    const char* WeBase = (const char*)We_b + ((size_t)wv << 12) + ((size_t)lane << 5);

    for (int kk = 0; kk < 16; ++kk) {
        const int k0 = kk << 5;
        FragU a[4];
#pragma unroll
        for (int mt = 0; mt < 4; ++mt) {
            const unsigned short* ap = &As[(mt * 16 + ll) * PS + k0 + hl * 8];
            a[mt].u.lo = *(const uint4*)ap;
            a[mt].u.hi = *(const uint4*)(ap + 16);
        }
        const char* wk = WeBase + ((size_t)kk << 15);
#pragma unroll
        for (int t = 0; t < 4; ++t) {
            FragU bb;
            bb.u.lo = *(const uint4*)(wk + (t << 10));
            bb.u.hi = *(const uint4*)(wk + (t << 10) + 16);
#pragma unroll
            for (int mt = 0; mt < 4; ++mt)
                acc[mt][t] = __builtin_amdgcn_wmma_f32_16x16x32_bf16(
                    false, a[mt].v, false, bb.v, (short)0, acc[mt][t], false, false);
        }
    }

    // phase 3: epilogue — tanh + dot with v_w, LDS reduction over N
    const float* qb = q + (size_t)b * HD;
    float qv[4], vw[4];
#pragma unroll
    for (int t = 0; t < 4; ++t) {
        const int n = (wv << 6) + (t << 4) + ll;
        qv[t] = qb[n];
        vw[t] = v_w[n];
    }
#pragma unroll
    for (int mt = 0; mt < 4; ++mt) {
        float part[8] = {};
#pragma unroll
        for (int t = 0; t < 4; ++t) {
#pragma unroll
            for (int r = 0; r < 8; ++r) {
                float x  = acc[mt][t][r] + qv[t];
                float e  = __builtin_amdgcn_exp2f(x * 2.88539008177793f); // exp(2x)
                float th = (e - 1.0f) * __builtin_amdgcn_rcpf(e + 1.0f);  // tanh(x)
                part[r] += th * vw[t];
            }
        }
#pragma unroll
        for (int r = 0; r < 8; ++r)       // D layout: lanes 16-31 hold M=8..15
            atomicAdd(&sred[mt * 16 + r + (hl << 3)], part[r]);
    }
    __syncthreads();
    if (tid < 64)
        out_scores[(size_t)b * SD + s0 + tid] = sred[tid] + v_b[0];
}

// ---------- kernel 4: in-place softmax over S per batch row ----------
__global__ __launch_bounds__(256) void softmax_kernel(float* __restrict__ out) {
    __shared__ float red[256];
    const int b = blockIdx.x, tid = threadIdx.x;
    float* row = out + (size_t)b * SD;

    float m = -3.4e38f;
#pragma unroll
    for (int j = 0; j < SD / 256; ++j) m = fmaxf(m, row[tid + j * 256]);
    red[tid] = m;
    __syncthreads();
    for (int off = 128; off > 0; off >>= 1) {
        if (tid < off) red[tid] = fmaxf(red[tid], red[tid + off]);
        __syncthreads();
    }
    m = red[0];
    __syncthreads();

    float s = 0.f, vals[SD / 256];
#pragma unroll
    for (int j = 0; j < SD / 256; ++j) {
        float e = __builtin_amdgcn_exp2f((row[tid + j * 256] - m) * 1.44269504088896f);
        vals[j] = e;
        s += e;
    }
    red[tid] = s;
    __syncthreads();
    for (int off = 128; off > 0; off >>= 1) {
        if (tid < off) red[tid] += red[tid + off];
        __syncthreads();
    }
    const float inv = 1.0f / red[0];
#pragma unroll
    for (int j = 0; j < SD / 256; ++j) row[tid + j * 256] = vals[j] * inv;
}

// --------------------------- launcher ---------------------------
extern "C" void kernel_launch(void* const* d_in, const int* in_sizes, int n_in,
                              void* d_out, int out_size, void* d_ws, size_t ws_size,
                              hipStream_t stream) {
    const float* hidden = (const float*)d_in[0];
    const float* enc    = (const float*)d_in[1];
    const float* W_att  = (const float*)d_in[2];
    const float* b_att  = (const float*)d_in[3];
    const float* v_w    = (const float*)d_in[4];
    const float* v_b    = (const float*)d_in[5];
    float* out = (float*)d_out;

    unsigned char* ws = (unsigned char*)d_ws;
    unsigned short* We_b = (unsigned short*)ws;                 // 512 KB packed bf16 W_e
    float* q             = (float*)(ws + (size_t)HD * HD * 2);  // 64 KB q

    we_convert<<<512, 256, 0, stream>>>(W_att, (unsigned*)We_b);
    q_kernel<<<(BD * HD) / 256, 256, 0, stream>>>(hidden, W_att, b_att, q);
    attn_gemm<<<(BD * SD) / 64, 256, 0, stream>>>(enc, We_b, q, v_w, v_b, out);
    softmax_kernel<<<BD, 256, 0, stream>>>(out);
}
